// MoECausalLM_61443802137314
// MI455X (gfx1250) — matrix operand order repaired and checked by reading, not compiled, not run on hardware
//
#include <hip/hip_runtime.h>
#include <math.h>

// Problem constants (match reference)
#define BT 2
#define ST 2048
#define HD 1024
#define FD 2816
#define NE 8
#define TK 2
#define NT (BT*ST)          // 4096 tokens

// Expert-kernel tiling (bf16 WMMA 16x16x32)
#define MT 64               // tokens per block (4 M-subtiles of 16)
#define FC 256              // F-chunk (16 waves x 16 f-cols)
#define XST 1032            // x-tile LDS row stride in bf16 elems (16B aligned rows)
#define HST 264             // h-tile LDS row stride in bf16 elems

typedef __attribute__((ext_vector_type(16))) __bf16 v16bf;
typedef __attribute__((ext_vector_type(8)))  float  v8f;

union U16frag { uint4 q[2]; v16bf v; };

// round-to-nearest-even f32 -> bf16 (as raw u16)
__device__ __forceinline__ unsigned short f2bf(float f) {
    unsigned u = __float_as_uint(f);
    u += 0x7fffu + ((u >> 16) & 1u);
    return (unsigned short)(u >> 16);
}

// 32 contiguous bytes -> B fragment (lane's K-half is contiguous in transposed weights)
__device__ __forceinline__ v16bf ld_b_frag(const unsigned short* p) {
    U16frag u;
    u.q[0] = *(const uint4*)(p);
    u.q[1] = *(const uint4*)(p + 8);
    return u.v;
}

// A fragment: elems 0..15 = K {base..base+15}
__device__ __forceinline__ v16bf ld_a_frag(const unsigned short* p) {
    U16frag u;
    u.q[0] = *(const uint4*)(p);
    u.q[1] = *(const uint4*)(p + 8);
    return u.v;
}

// ---------------------------------------------------------------------------
// Zero the small aux accumulators in workspace
// ---------------------------------------------------------------------------
__global__ void moe_zero_kernel(float* __restrict__ acc) {
    if (threadIdx.x < 16) acc[threadIdx.x] = 0.0f;
}

// ---------------------------------------------------------------------------
// f32 -> bf16 convert + transpose: src[R][C] -> dst[C][R], one expert per z
// ---------------------------------------------------------------------------
__global__ __launch_bounds__(256) void transpose_bf16_kernel(
    const float* __restrict__ src, unsigned short* __restrict__ dst, int R, int C)
{
    __shared__ float tile[32][33];
    const size_t eoff = (size_t)blockIdx.z * R * C;
    src += eoff; dst += eoff;
    const int c0 = blockIdx.x * 32, r0 = blockIdx.y * 32;
    const int tx = threadIdx.x, ty = threadIdx.y;
    #pragma unroll
    for (int i = ty; i < 32; i += 8)
        tile[i][tx] = src[(size_t)(r0 + i) * C + c0 + tx];
    __syncthreads();
    #pragma unroll
    for (int i = ty; i < 32; i += 8)
        dst[(size_t)(c0 + i) * R + r0 + tx] = f2bf(tile[tx][i]);
}

// ---------------------------------------------------------------------------
// Router: logits, top-2 softmax combine weights, aux-loss accumulators
// One wave32 per token; 8 waves per block.
// ---------------------------------------------------------------------------
__global__ __launch_bounds__(256) void moe_router_kernel(
    const float* __restrict__ x, const float* __restrict__ rw,
    float* __restrict__ wcomb, float* __restrict__ Pacc, float* __restrict__ facc)
{
    const int wave = threadIdx.x >> 5;
    const int lane = threadIdx.x & 31;
    const int t = blockIdx.x * 8 + wave;
    if (t >= NT) return;

    const float* xr = x + (size_t)t * HD;
    float logits[NE];
    #pragma unroll
    for (int e = 0; e < NE; ++e) {
        float acc = 0.0f;
        const float* wr = rw + e * HD;
        for (int h = lane; h < HD; h += 32) acc += xr[h] * wr[h];
        #pragma unroll
        for (int off = 16; off > 0; off >>= 1) acc += __shfl_xor(acc, off, 32);
        logits[e] = acc;   // identical across all lanes
    }

    int i0 = 0; float v0 = logits[0];
    #pragma unroll
    for (int e = 1; e < NE; ++e) if (logits[e] > v0) { v0 = logits[e]; i0 = e; }
    int i1 = (i0 == 0) ? 1 : 0; float v1 = logits[i1];
    #pragma unroll
    for (int e = 0; e < NE; ++e)
        if (e != i0 && logits[e] > v1) { v1 = logits[e]; i1 = e; }

    const float ew = __expf(v1 - v0);
    const float w0 = 1.0f / (1.0f + ew);
    const float w1 = ew / (1.0f + ew);

    if (lane < NE) {
        const float w = (lane == i0) ? w0 : ((lane == i1) ? w1 : 0.0f);
        wcomb[(size_t)t * NE + lane] = w;
        float s = 0.0f;
        #pragma unroll
        for (int e = 0; e < NE; ++e) s += __expf(logits[e] - v0);
        atomicAdd(&Pacc[lane], __expf(logits[lane] - v0) / s);
    }
    if (lane == 0) {
        atomicAdd(&facc[i0], 1.0f);
        atomicAdd(&facc[i1], 1.0f);
    }
}

// ---------------------------------------------------------------------------
// Fused dense MoE expert MLP, bf16 WMMA (v_wmma_f32_16x16x32_bf16, f32 accum).
// Block = 512 threads (16 waves) owns 64 tokens. Experts looped inside so the
// routing-weighted sum accumulates in registers (h pre-scaled by w_e).
// ---------------------------------------------------------------------------
__global__ __launch_bounds__(512) void moe_expert_kernel(
    const float* __restrict__ x,
    const unsigned short* __restrict__ wgT,   // [E][F][H] bf16 (transposed)
    const unsigned short* __restrict__ wuT,   // [E][F][H] bf16
    const unsigned short* __restrict__ wdT,   // [E][H][F] bf16 (transposed)
    const float* __restrict__ wcomb,
    float* __restrict__ out)
{
    extern __shared__ unsigned short smem_u16[];
    unsigned short* xs = smem_u16;                // MT x XST bf16
    unsigned short* hs = xs + MT * XST;           // MT x HST bf16
    float* wl = (float*)(hs + MT * HST);          // MT combine weights

    const int tid   = threadIdx.x;
    const int wave  = tid >> 5;
    const int lane  = tid & 31;
    const int lhalf = lane >> 4;
    const int lm    = lane & 15;
    const int t0    = blockIdx.x * MT;

    // Stage x tile (64 x 1024) into LDS as bf16
    for (int i = tid; i < MT * (HD / 4); i += 512) {
        const int r  = i >> 8;            // HD/4 == 256
        const int c  = (i & 255) << 2;
        const float4 v = *(const float4*)(x + (size_t)(t0 + r) * HD + c);
        const unsigned p0 = (unsigned)f2bf(v.x) | ((unsigned)f2bf(v.y) << 16);
        const unsigned p1 = (unsigned)f2bf(v.z) | ((unsigned)f2bf(v.w) << 16);
        *(uint2*)(xs + r * XST + c) = make_uint2(p0, p1);
    }
    __syncthreads();

    // Persistent weighted output accumulators: 64 tokens x 64 H-cols per wave
    v8f out_acc[16];   // [msub(4)][ntile(4)]
    #pragma unroll
    for (int i = 0; i < 16; ++i) out_acc[i] = (v8f){};

    for (int e = 0; e < NE; ++e) {
        const unsigned short* wgE = wgT + (size_t)e * FD * HD;
        const unsigned short* wuE = wuT + (size_t)e * FD * HD;
        const unsigned short* wdE = wdT + (size_t)e * HD * FD;

        if (tid < MT) wl[tid] = wcomb[(size_t)(t0 + tid) * NE + e];
        __syncthreads();

        for (int fc = 0; fc < FD; fc += FC) {
            // ---- Phase A: gate/up for wave's 16 F-columns over K = H ----
            v8f ga[4], ua[4];
            #pragma unroll
            for (int i = 0; i < 4; ++i) { ga[i] = (v8f){}; ua[i] = (v8f){}; }
            const int col = fc + wave * 16 + lm;
            const unsigned short* pg = wgE + (size_t)col * HD + lhalf * 16;
            const unsigned short* pu = wuE + (size_t)col * HD + lhalf * 16;
            #pragma unroll 1
            for (int k = 0; k < HD; k += 32) {
                const v16bf bg = ld_b_frag(pg + k);
                const v16bf bu = ld_b_frag(pu + k);
                #pragma unroll
                for (int ms = 0; ms < 4; ++ms) {
                    const v16bf a =
                        ld_a_frag(xs + (ms * 16 + lm) * XST + k + lhalf * 16);
                    ga[ms] = __builtin_amdgcn_wmma_f32_16x16x32_bf16(false, a, false, bg, (short)0, ga[ms], false, false);
                    ua[ms] = __builtin_amdgcn_wmma_f32_16x16x32_bf16(false, a, false, bu, (short)0, ua[ms], false, false);
                }
            }

            __syncthreads();   // previous chunk's down-proj done reading hs

            // silu(g)*u, pre-scaled by token combine weight -> hs (bf16)
            #pragma unroll
            for (int ms = 0; ms < 4; ++ms) {
                #pragma unroll
                for (int j = 0; j < 8; ++j) {
                    const int m  = ms * 16 + j + (lhalf << 3);
                    const float gv = ga[ms][j];
                    const float sv = gv * __builtin_amdgcn_rcpf(1.0f + __expf(-gv));
                    hs[m * HST + wave * 16 + lm] = f2bf(sv * ua[ms][j] * wl[m]);
                }
            }
            __syncthreads();

            // ---- Phase B: down-proj, wave owns H-cols [wave*64, +64) ----
            const int ch = wave * 64;
            #pragma unroll 1
            for (int kk = 0; kk < FC; kk += 32) {
                v16bf a[4];
                #pragma unroll
                for (int ms = 0; ms < 4; ++ms)
                    a[ms] = ld_a_frag(hs + (ms * 16 + lm) * HST + kk + lhalf * 16);
                #pragma unroll
                for (int nt = 0; nt < 4; ++nt) {
                    const v16bf b = ld_b_frag(
                        wdE + (size_t)(ch + nt * 16 + lm) * FD + fc + kk + lhalf * 16);
                    #pragma unroll
                    for (int ms = 0; ms < 4; ++ms)
                        out_acc[ms * 4 + nt] = __builtin_amdgcn_wmma_f32_16x16x32_bf16(false, a[ms], false, b, (short)0, out_acc[ms * 4 + nt], false, false);
                }
            }
        }
        __syncthreads();
    }

    // Final store (blocks own disjoint tokens -> plain stores)
    #pragma unroll
    for (int ms = 0; ms < 4; ++ms) {
        #pragma unroll
        for (int nt = 0; nt < 4; ++nt) {
            const v8f acc = out_acc[ms * 4 + nt];
            #pragma unroll
            for (int j = 0; j < 8; ++j) {
                const int m = ms * 16 + j + (lhalf << 3);
                const int c = wave * 64 + nt * 16 + lm;
                out[(size_t)(t0 + m) * HD + c] = acc[j];
            }
        }
    }
}

// ---------------------------------------------------------------------------
// Aux loss finalize: aux = E * sum_e f[e] * P[e]
// ---------------------------------------------------------------------------
__global__ void moe_aux_kernel(const float* __restrict__ Pacc,
                               const float* __restrict__ facc,
                               float* __restrict__ out_aux)
{
    if (threadIdx.x == 0) {
        float s = 0.0f;
        #pragma unroll
        for (int e = 0; e < NE; ++e)
            s += (facc[e] / (float)(NT * TK)) * (Pacc[e] / (float)NT);
        *out_aux = (float)NE * s;
    }
}

// ---------------------------------------------------------------------------
extern "C" void kernel_launch(void* const* d_in, const int* in_sizes, int n_in,
                              void* d_out, int out_size, void* d_ws, size_t ws_size,
                              hipStream_t stream) {
    const float* x   = (const float*)d_in[0];   // [T,H]
    const float* rw  = (const float*)d_in[1];   // [E,H]
    const float* wg  = (const float*)d_in[2];   // [E,H,F]
    const float* wu  = (const float*)d_in[3];   // [E,H,F]
    const float* wd  = (const float*)d_in[4];   // [E,F,H]
    float* out = (float*)d_out;                 // [T,H] + 1 aux

    // Workspace layout
    float* wsf   = (float*)d_ws;
    float* Pacc  = wsf;                         // 8
    float* facc  = wsf + 8;                     // 8
    float* wcomb = wsf + 16;                    // T*E f32
    unsigned short* wgT = (unsigned short*)(wcomb + (size_t)NT * NE); // [E][F][H]
    unsigned short* wuT = wgT + (size_t)NE * HD * FD;                 // [E][F][H]
    unsigned short* wdT = wuT + (size_t)NE * HD * FD;                 // [E][H][F]

    moe_zero_kernel<<<1, 32, 0, stream>>>(wsf);

    // f32 -> bf16 transposed weight copies (one-shot streaming pass)
    const dim3 tb(32, 8);
    transpose_bf16_kernel<<<dim3(FD / 32, HD / 32, NE), tb, 0, stream>>>(wg, wgT, HD, FD);
    transpose_bf16_kernel<<<dim3(FD / 32, HD / 32, NE), tb, 0, stream>>>(wu, wuT, HD, FD);
    transpose_bf16_kernel<<<dim3(HD / 32, FD / 32, NE), tb, 0, stream>>>(wd, wdT, FD, HD);

    moe_router_kernel<<<NT / 8, 256, 0, stream>>>(x, rw, wcomb, Pacc, facc);

    const size_t lds_bytes =
        (size_t)(MT * XST + MT * HST) * sizeof(unsigned short) + MT * sizeof(float);
    moe_expert_kernel<<<NT / MT, 512, lds_bytes, stream>>>(x, wgT, wuT, wdT, wcomb, out);

    moe_aux_kernel<<<1, 32, 0, stream>>>(Pacc, facc, out + (size_t)NT * HD);
}
